// UResNet_80298708566481
// MI455X (gfx1250) — matrix operand order repaired and checked by reading, not compiled, not run on hardware
//
#include <hip/hip_runtime.h>
#include <hip/hip_bf16.h>

typedef _Float16 f16;
typedef __attribute__((ext_vector_type(16))) _Float16      v16h;
typedef __attribute__((ext_vector_type(8)))  float         v8f;
typedef __attribute__((ext_vector_type(8)))  unsigned int  v8u;
typedef __attribute__((ext_vector_type(4)))  unsigned int  u32x4;
typedef __attribute__((ext_vector_type(8)))  int           i32x8;
typedef __attribute__((ext_vector_type(4)))  int           i32x4;

#define DVOL 96
#define EPSV 1e-4f

// ---------------------------------------------------------------- utility kernels
__global__ void k_zero(float* p, long n) {
  long i = (long)blockIdx.x * blockDim.x + threadIdx.x;
  if (i < n) p[i] = 0.f;
}

__global__ void k_scatter(const int* __restrict__ coords, const float* __restrict__ feats,
                          float* __restrict__ xd, float* __restrict__ m0, int npts) {
  int p = blockIdx.x * blockDim.x + threadIdx.x;
  if (p >= npts) return;
  int cx = coords[p*3+0], cy = coords[p*3+1], cz = coords[p*3+2];
  int v = (cx * DVOL + cy) * DVOL + cz;
  xd[v] = feats[p];
  m0[v] = 1.f;
}

__global__ void k_pool(const float* __restrict__ mi, float* __restrict__ mo, int Do) {
  int i = blockIdx.x * blockDim.x + threadIdx.x;
  int n = Do * Do * Do;
  if (i >= n) return;
  int z = i / (Do * Do), r = i % (Do * Do), y = r / Do, x = r % Do;
  int Di = Do * 2;
  float m = 0.f;
  for (int dz = 0; dz < 2; ++dz)
    for (int dy = 0; dy < 2; ++dy)
      for (int dx = 0; dx < 2; ++dx)
        m = fmaxf(m, mi[((z*2+dz) * Di + (y*2+dy)) * Di + (x*2+dx)]);
  mo[i] = m;
}

__global__ void k_stem(const float* __restrict__ xd, const float* __restrict__ w,
                       const float* __restrict__ m0, float* __restrict__ out) {
  int v = blockIdx.x * blockDim.x + threadIdx.x;
  int n = DVOL * DVOL * DVOL;
  if (v >= n) return;
  int z = v / (DVOL*DVOL), r = v % (DVOL*DVOL), y = r / DVOL, x = r % DVOL;
  float acc[16];
#pragma unroll
  for (int c = 0; c < 16; ++c) acc[c] = 0.f;
  for (int t = 0; t < 27; ++t) {
    int dz = t/9 - 1, dy = (t/3)%3 - 1, dx = t%3 - 1;
    int zi = z+dz, yi = y+dy, xi = x+dx;
    if ((unsigned)zi >= DVOL || (unsigned)yi >= DVOL || (unsigned)xi >= DVOL) continue;
    float xv = xd[(zi*DVOL + yi)*DVOL + xi];
    if (xv != 0.f) {
#pragma unroll
      for (int c = 0; c < 16; ++c) acc[c] += xv * w[t*16 + c];
    }
  }
  float m = m0[v];
#pragma unroll
  for (int c = 0; c < 16; ++c) out[(long)v*16 + c] = acc[c] * m;
}

// masked BN stats: stats[0..C-1]=sum(x*m), stats[C..2C-1]=sum((x*m)^2), stats[2C]=sum(m)
__global__ void k_bn_stats(const float* __restrict__ x, const float* __restrict__ mask,
                           float* __restrict__ stats, int C, int nvox) {
  __shared__ float sh[2*64 + 1];
  int tid = threadIdx.x;
  if (tid < 2*C + 1) sh[tid] = 0.f;
  __syncthreads();
  int c = tid % C, row = tid / C, rows = blockDim.x / C;
  float s = 0.f, ss = 0.f, cnt = 0.f;
  for (int v = blockIdx.x * rows + row; v < nvox; v += gridDim.x * rows) {
    float m = mask[v];
    float val = x[(long)v * C + c] * m;
    s += val; ss += val * val;
    if (c == 0) cnt += m;
  }
  atomicAdd(&sh[c], s);
  atomicAdd(&sh[C + c], ss);
  if (c == 0) atomicAdd(&sh[2*C], cnt);
  __syncthreads();
  if (tid < 2*C + 1) atomicAdd(&stats[tid], sh[tid]);
}

__global__ void k_bn_apply(const float* __restrict__ x, const float* __restrict__ mask,
                           const float* __restrict__ stats, const float* __restrict__ g,
                           const float* __restrict__ b, f16* __restrict__ o16,
                           float* __restrict__ o32, int C, long n) {
  long i = (long)blockIdx.x * blockDim.x + threadIdx.x;
  if (i >= n) return;
  int c = (int)(i % C);
  long v = i / C;
  float cnt  = stats[2*C];
  float mean = stats[c] / cnt;
  float var  = stats[C + c] / cnt - mean * mean;
  float y = g[c] * (x[i] - mean) * rsqrtf(var + EPSV) + b[c];
  y = fmaxf(y, 0.f) * mask[v];
  o16[i] = (f16)y;
  if (o32) o32[i] = y;
}

// pack conv weight [t][ci][co] f32 -> co-major f16: wp[co][t*cin + ci]
__global__ void k_pack_w(const float* __restrict__ w, f16* __restrict__ wp,
                         int ntaps, int cin, int cout) {
  int i = blockIdx.x * blockDim.x + threadIdx.x;
  int n = ntaps * cin * cout;
  if (i >= n) return;
  int t = i / (cin * cout), r = i % (cin * cout), ci = r / cout, co = r % cout;
  wp[(long)co * (ntaps * cin) + t * cin + ci] = (f16)w[i];
}

// ---------------------------------------------------------------- implicit-GEMM conv (WMMA)
template<int TAPDIM>
__device__ __forceinline__ void tap_decode(int t, int& dz, int& dy, int& dx) {
  if (TAPDIM == 3)      { dz = t/9 - 1; dy = (t/3)%3 - 1; dx = t%3 - 1; }
  else if (TAPDIM == 2) { dz = (t>>2)&1; dy = (t>>1)&1;  dx = t&1; }
  else                  { dz = 0; dy = 0; dx = 0; }
}

// act: f16 dense activations [DIN^3][CIN]; wpk: f16 weights co-major [COUT][NTAPS*CIN]
// out[vox][co] = WMMA-GEMM * mask[vox] (+ sc[vox][co])
template<int CIN, int COUT, int TAPDIM, int STRIDE, int DIN, int DOUT>
__global__ __launch_bounds__(256)
void k_conv_wmma(const f16* __restrict__ act, const f16* __restrict__ wpk,
                 const float* __restrict__ mask, const float* __restrict__ sc,
                 float* __restrict__ out) {
  constexpr int NTAPS = (TAPDIM == 3) ? 27 : ((TAPDIM == 2) ? 8 : 1);
  constexpr int KT    = NTAPS * CIN;            // multiple of 16 for all combos used
  constexpr int NCH   = (KT + 31) / 32;
  constexpr int NT    = COUT / 16;
  constexpr int D2    = DOUT * DOUT;

  __shared__ __align__(16) f16 ldsB[16 * KT];   // one N-tile of B per block

  int ntile = (NT == 1) ? 0 : ((int)blockIdx.x % NT);
  int mgrp  = (NT == 1) ? (int)blockIdx.x : ((int)blockIdx.x / NT);
  int coBase = ntile * 16;

  // ---- stage B tile (16 rows x KT f16, contiguous in co-major layout) into LDS ----
#if __has_builtin(__builtin_amdgcn_tensor_load_to_lds)
  if (threadIdx.x < 32) {                       // single wave issues the TDM op
    unsigned long long ga = (unsigned long long)(const void*)(wpk + (long)coBase * KT);
    unsigned ldsAddr = (unsigned)(unsigned long long)(void*)ldsB;
    unsigned td0 = (unsigned)(16 * KT);         // 1-D tile, data_size = 2 bytes
    u32x4 g0; i32x8 g1; i32x4 g2, g3; i32x8 g4;
    g0[0] = 1u;                                 // count=1, user descriptor
    g0[1] = ldsAddr;                            // lds_addr
    g0[2] = (unsigned)(ga & 0xffffffffu);       // global_addr lo
    g0[3] = (unsigned)((ga >> 32) & 0x01ffffffu) | (2u << 30);  // addr hi | type=2
    g1[0] = (int)0x00010000u;                   // data_size=1 (2B), no multicast
    g1[1] = (int)((td0 & 0xffffu) << 16);       // tensor_dim0 lo16
    g1[2] = (int)(((td0 >> 16) & 0xffffu) | (1u << 16)); // dim0 hi | tensor_dim1=1
    g1[3] = (int)((td0 & 0xffffu) << 16);       // tile_dim0
    g1[4] = 1;                                  // tile_dim1 = 1
    g1[5] = (int)td0;                           // tensor_dim0_stride lo
    g1[6] = 0; g1[7] = 0;
    g2[0] = 0; g2[1] = 0; g2[2] = 0; g2[3] = 0;
    g3[0] = 0; g3[1] = 0; g3[2] = 0; g3[3] = 0;
    g4[0] = 0; g4[1] = 0; g4[2] = 0; g4[3] = 0;
    g4[4] = 0; g4[5] = 0; g4[6] = 0; g4[7] = 0;
    __builtin_amdgcn_tensor_load_to_lds(g0, g1, g2, g3, g4, 0);  // 6-arg toolchain form
#if __has_builtin(__builtin_amdgcn_s_wait_tensorcnt)
    __builtin_amdgcn_s_wait_tensorcnt(0);
#else
    asm volatile("s_wait_tensorcnt 0x0" ::: "memory");
#endif
  }
#else
  {
    const unsigned* wpkU = (const unsigned*)wpk;
    unsigned* ldsU = (unsigned*)ldsB;
    for (int u = threadIdx.x; u < KT * 8; u += 256)
      ldsU[u] = wpkU[coBase * (KT / 2) + u];
  }
#endif
  __syncthreads();

  int lane = threadIdx.x & 31;
  int mtile = mgrp * 8 + (int)(threadIdx.x >> 5);
  int m  = lane & 15;                           // A row / B column owned by lane
  int kh = lane >> 4;                           // lane half -> K sub-range
  int voxA = mtile * 16 + m;
  int zo = voxA / D2, rem = voxA % D2, yo = rem / DOUT, xo = rem % DOUT;

  const u32x4* actV = (const u32x4*)act;
  const u32x4* ldsV = (const u32x4*)ldsB;
  __builtin_prefetch(act + (long)voxA * CIN, 0, 0);   // global_prefetch_b8

  v8f acc[2];
#pragma unroll
  for (int j = 0; j < 8; ++j) { acc[0][j] = 0.f; acc[1][j] = 0.f; }

#pragma unroll
  for (int ic = 0; ic < NCH; ++ic) {
    const int kk = ic * 32;
    v8u au, bu;
    // ---- A: two 8-wide K runs (contiguous f16 -> b128 loads) ----
#pragma unroll
    for (int run = 0; run < 2; ++run) {
      const int kblk = kk + run * 16;           // 16-aligned, stays inside one tap
      u32x4 r; r[0] = 0u; r[1] = 0u; r[2] = 0u; r[3] = 0u;
      if (kblk < KT) {
        const int t = kblk / CIN;               // compile-time after unroll
        int dz, dy, dx; tap_decode<TAPDIM>(t, dz, dy, dx);
        int zi = zo * STRIDE + dz, yi = yo * STRIDE + dy, xi = xo * STRIDE + dx;
        bool ok = (TAPDIM != 3) ||
                  ((unsigned)zi < (unsigned)DIN && (unsigned)yi < (unsigned)DIN &&
                   (unsigned)xi < (unsigned)DIN);
        if (ok) {
          int cs = (kblk % CIN) + kh * 8;       // channel start, multiple of 8
          r = actV[(((zi * DIN + yi) * DIN + xi) * CIN + cs) >> 3];
        }
      }
      au[run*4+0] = r[0]; au[run*4+1] = r[1]; au[run*4+2] = r[2]; au[run*4+3] = r[3];
    }
    // ---- B: two 8-wide K runs 16 apart from LDS (2 x ds_load_b128) ----
    {
      const int kb = kk + kh * 8;
      u32x4 l0; l0[0]=0u; l0[1]=0u; l0[2]=0u; l0[3]=0u;
      u32x4 l1 = l0;
      if (kb < KT) {
        int bi = (m * KT + kb) >> 3;
        l0 = ldsV[bi];
        if (kb + 16 < KT) l1 = ldsV[bi + 2];
      }
      bu[0] = l0[0]; bu[1] = l0[1]; bu[2] = l0[2]; bu[3] = l0[3];
      bu[4] = l1[0]; bu[5] = l1[1]; bu[6] = l1[2]; bu[7] = l1[3];
    }
    v16h a = __builtin_bit_cast(v16h, au);
    v16h b = __builtin_bit_cast(v16h, bu);
    acc[ic & 1] = __builtin_amdgcn_wmma_f32_16x16x32_f16(false, a, false, b, (short)0,
                                                         acc[ic & 1], false, false);
  }

  // D layout: VGPR r -> row M = r + kh*8, column N = lane%16
  int co = coBase + m;
#pragma unroll
  for (int r = 0; r < 8; ++r) {
    int vox = mtile * 16 + kh * 8 + r;
    float val = (acc[0][r] + acc[1][r]) * mask[vox];
    if (sc) val += sc[(long)vox * COUT + co];
    out[(long)vox * COUT + co] = val;
  }
}

// transpose conv 2x2x2 stride2: each output voxel has exactly one (parity) tap
__global__ void k_upconv(const f16* __restrict__ act, const float* __restrict__ w,
                         const float* __restrict__ mask, float* __restrict__ out,
                         int Din, int Dout, int B, int A) {
  int i = blockIdx.x * blockDim.x + threadIdx.x;
  int n = Dout * Dout * Dout * A;
  if (i >= n) return;
  int co = i % A, vox = i / A;
  int D2 = Dout * Dout;
  int z = vox / D2, r = vox % D2, y = r / Dout, x = r % Dout;
  int tap = ((z & 1) * 2 + (y & 1)) * 2 + (x & 1);
  int iv = ((z >> 1) * Din + (y >> 1)) * Din + (x >> 1);
  const f16* ain = act + (long)iv * B;
  const float* wt = w + (long)tap * B * A + co;
  float s = 0.f;
  for (int cb = 0; cb < B; ++cb) s += (float)ain[cb] * wt[cb * A];
  out[i] = s * mask[vox];
}

__global__ void k_concat(const float* __restrict__ a, const float* __restrict__ bsrc,
                         float* __restrict__ o, f16* __restrict__ o16, int A, long nvox) {
  long i = (long)blockIdx.x * blockDim.x + threadIdx.x;
  long n = nvox * 2 * A;
  if (i >= n) return;
  int c = (int)(i % (2 * A));
  long v = i / (2 * A);
  float val = (c < A) ? a[v * A + c] : bsrc[v * A + (c - A)];
  o[i] = val;
  o16[i] = (f16)val;
}

__global__ void k_gather_lin(const int* __restrict__ coords, const float* __restrict__ x,
                             const float* __restrict__ lw, const float* __restrict__ lb,
                             float* __restrict__ out, int npts) {
  int p = blockIdx.x * blockDim.x + threadIdx.x;
  if (p >= npts) return;
  int cx = coords[p*3], cy = coords[p*3+1], cz = coords[p*3+2];
  int v = (cx * DVOL + cy) * DVOL + cz;
  float o0 = lb[0], o1 = lb[1], o2 = lb[2];
  for (int c = 0; c < 16; ++c) {
    float f = x[(long)v * 16 + c];
    o0 += f * lw[c*3+0]; o1 += f * lw[c*3+1]; o2 += f * lw[c*3+2];
  }
  out[p*3+0] = o0; out[p*3+1] = o1; out[p*3+2] = o2;
}

// ---------------------------------------------------------------- host orchestration
static inline int cdiv(long a, long b) { return (int)((a + b - 1) / b); }

template<int CIN, int COUT, int TAPDIM, int STRIDE, int DIN, int DOUT>
static void conv_launch(const f16* act, const f16* wpk, const float* mask,
                        const float* sc, float* out, hipStream_t s) {
  constexpr int blocks = (DOUT * DOUT * DOUT / 128) * (COUT / 16);
  hipLaunchKernelGGL((k_conv_wmma<CIN, COUT, TAPDIM, STRIDE, DIN, DOUT>),
                     dim3(blocks), dim3(256), 0, s, act, wpk, mask, sc, out);
}

static void bn_launch(const float* x, const float* mask, const float* g, const float* b,
                      f16* o16, float* o32, int C, long nvox, float* ST, hipStream_t s) {
  k_zero<<<1, 256, 0, s>>>(ST, 2*C + 1);
  k_bn_stats<<<512, C*4, 0, s>>>(x, mask, ST, C, (int)nvox);
  k_bn_apply<<<cdiv(nvox*C, 256), 256, 0, s>>>(x, mask, ST, g, b, o16, o32, C, nvox*C);
}

static void pack_launch(const float* w, f16* dst, int taps, int ci, int co, hipStream_t s) {
  int n = taps * ci * co;
  k_pack_w<<<cdiv(n, 256), 256, 0, s>>>(w, dst, taps, ci, co);
}

// residual block: X (A ch) -> OUT (B ch). pbase: b1,b2,g1,g2,[nin],w1,w2 (sorted pytree)
template<int A, int B, int DM, bool NIN>
static void resblock(void* const* d_in, int pbase, float* X, const f16* X16,
                     const float* mask, float* H, f16* Th, float* SC, float* OUT,
                     float* ST, f16* PKA, f16* PKB, hipStream_t stream) {
  auto F = [&](int i) { return (const float*)d_in[i]; };
  constexpr long nvox = (long)DM * DM * DM;
  const float* scp = X;
  if constexpr (NIN) {
    pack_launch(F(pbase+4), PKB, 1, A, B, stream);
    conv_launch<A, B, 1, 1, DM, DM>(X16, PKB, mask, nullptr, SC, stream);
    scp = SC;
  }
  bn_launch(X, mask, F(pbase+2), F(pbase+0), Th, nullptr, A, nvox, ST, stream);
  pack_launch(F(pbase + (NIN ? 5 : 4)), PKA, 27, A, B, stream);
  conv_launch<A, B, 3, 1, DM, DM>(Th, PKA, mask, nullptr, H, stream);
  bn_launch(H, mask, F(pbase+3), F(pbase+1), Th, nullptr, B, nvox, ST, stream);
  pack_launch(F(pbase + (NIN ? 6 : 5)), PKA, 27, B, B, stream);
  conv_launch<B, B, 3, 1, DM, DM>(Th, PKA, mask, scp, OUT, stream);
}

extern "C" void kernel_launch(void* const* d_in, const int* in_sizes, int n_in,
                              void* d_out, int out_size, void* d_ws, size_t ws_size,
                              hipStream_t stream) {
  if (n_in < 81) return;
  const int npts = in_sizes[1];
  const int* coords = (const int*)d_in[0];
  const float* feats = (const float*)d_in[1];
  auto F = [&](int i) { return (const float*)d_in[i]; };

  char* base = (char*)d_ws;
  size_t off = 0;
  auto alloc = [&](size_t bytes) -> void* {
    void* p = base + off;
    off += (bytes + 255) & ~(size_t)255;
    return p;
  };
  const long N0 = 96L*96*96, N1 = 48L*48*48, N2 = 24L*24*24;
  float* m0 = (float*)alloc(N0*4);  float* m1 = (float*)alloc(N1*4);  float* m2 = (float*)alloc(N2*4);
  float* xd = (float*)alloc(N0*4);
  float* X0 = (float*)alloc(N0*16*4); float* H0 = (float*)alloc(N0*16*4);
  float* SC0 = (float*)alloc(N0*16*4); float* Y0 = (float*)alloc(N0*16*4);
  float* C0 = (float*)alloc(N0*32*4); float* F0 = (float*)alloc(N0*16*4);
  f16* T0h = (f16*)alloc(N0*32*2);  f16* C0h = (f16*)alloc(N0*32*2);
  float* X1 = (float*)alloc(N1*32*4); float* H1 = (float*)alloc(N1*32*4);
  float* SC1 = (float*)alloc(N1*32*4); float* Y1 = (float*)alloc(N1*32*4);
  float* C1 = (float*)alloc(N1*64*4);
  f16* T1h = (f16*)alloc(N1*64*2);  f16* C1h = (f16*)alloc(N1*64*2);
  float* X2 = (float*)alloc(N2*48*4); float* H2 = (float*)alloc(N2*48*4);
  f16* T2h = (f16*)alloc(N2*48*2);
  f16* PKA = (f16*)alloc(70000*2);  f16* PKB = (f16*)alloc(70000*2);
  float* ST = (float*)alloc(256*4);

  auto zero = [&](float* p, long n) { k_zero<<<cdiv(n,256), 256, 0, stream>>>(p, n); };

  // ---- scatter + masks ----
  zero(xd, N0); zero(m0, N0);
  k_scatter<<<cdiv(npts, 256), 256, 0, stream>>>(coords, feats, xd, m0, npts);
  k_pool<<<cdiv(N1, 256), 256, 0, stream>>>(m0, m1, 48);
  k_pool<<<cdiv(N2, 256), 256, 0, stream>>>(m1, m2, 24);
  // ---- stem ----
  k_stem<<<cdiv(N0, 256), 256, 0, stream>>>(xd, F(6), m0, X0);
  // ---- level0 pre ----
  resblock<16,16,96,false>(d_in, 66, X0, nullptr, m0, H0, T0h, SC0, X0, ST, PKA, PKB, stream);
  resblock<16,16,96,false>(d_in, 72, X0, nullptr, m0, H0, T0h, SC0, X0, ST, PKA, PKB, stream);
  // ---- down 0 (16 -> 32, 96 -> 48) ----
  bn_launch(X0, m0, F(8), F(7), T0h, nullptr, 16, N0, ST, stream);
  pack_launch(F(9), PKA, 8, 16, 32, stream);
  conv_launch<16,32,2,2,96,48>(T0h, PKA, m1, nullptr, X1, stream);
  // ---- level1 pre ----
  resblock<32,32,48,false>(d_in, 38, X1, nullptr, m1, H1, T1h, SC1, X1, ST, PKA, PKB, stream);
  resblock<32,32,48,false>(d_in, 44, X1, nullptr, m1, H1, T1h, SC1, X1, ST, PKA, PKB, stream);
  // ---- down 1 (32 -> 48, 48 -> 24) ----
  bn_launch(X1, m1, F(11), F(10), T1h, nullptr, 32, N1, ST, stream);
  pack_launch(F(12), PKA, 8, 32, 48, stream);
  conv_launch<32,48,2,2,48,24>(T1h, PKA, m2, nullptr, X2, stream);
  // ---- level2 (innermost) ----
  resblock<48,48,24,false>(d_in, 13, X2, nullptr, m2, H2, T2h, nullptr, X2, ST, PKA, PKB, stream);
  resblock<48,48,24,false>(d_in, 19, X2, nullptr, m2, H2, T2h, nullptr, X2, ST, PKA, PKB, stream);
  // ---- up to level1 (48 -> 32, 24 -> 48) ----
  bn_launch(X2, m2, F(51), F(50), T2h, nullptr, 48, N2, ST, stream);
  k_upconv<<<cdiv(N1*32, 256), 256, 0, stream>>>(T2h, F(52), m1, Y1, 24, 48, 48, 32);
  k_concat<<<cdiv(N1*64, 256), 256, 0, stream>>>(X1, Y1, C1, C1h, 32, N1);
  // ---- level1 post ----
  resblock<64,32,48,true >(d_in, 25, C1, C1h, m1, H1, T1h, SC1, X1, ST, PKA, PKB, stream);
  resblock<32,32,48,false>(d_in, 32, X1, nullptr, m1, H1, T1h, SC1, X1, ST, PKA, PKB, stream);
  // ---- up to level0 (32 -> 16, 48 -> 96) ----
  bn_launch(X1, m1, F(79), F(78), T1h, nullptr, 32, N1, ST, stream);
  k_upconv<<<cdiv(N0*16, 256), 256, 0, stream>>>(T1h, F(80), m0, Y0, 48, 96, 32, 16);
  k_concat<<<cdiv(N0*32, 256), 256, 0, stream>>>(X0, Y0, C0, C0h, 16, N0);
  // ---- level0 post ----
  resblock<32,16,96,true >(d_in, 53, C0, C0h, m0, H0, T0h, SC0, X0, ST, PKA, PKB, stream);
  resblock<16,16,96,false>(d_in, 60, X0, nullptr, m0, H0, T0h, SC0, X0, ST, PKA, PKB, stream);
  // ---- final BN + gather + linear ----
  bn_launch(X0, m0, F(3), F(2), T0h, F0, 16, N0, ST, stream);
  k_gather_lin<<<cdiv(npts, 256), 256, 0, stream>>>(coords, F0, F(5), F(4),
                                                    (float*)d_out, npts);
}
